// DynamicMultiHeadAttention_71932112273573
// MI455X (gfx1250) — matrix operand order repaired and checked by reading, not compiled, not run on hardware
//
#include <hip/hip_runtime.h>

typedef __bf16 bf16_t;
typedef __attribute__((ext_vector_type(16))) __bf16 v16bf;
typedef __attribute__((ext_vector_type(8)))  float  v8f;

#define B_SZ   2
#define S_LEN  2048
#define D_DIM  1024
#define H_NUM  8
#define HD_DIM 128
#define LDV    40   // padded row length (elements) for transposed V tile; 80B keeps 16B align

static __device__ __forceinline__ v8f wmma_bf16(v16bf a, v16bf b, v8f c) {
    // D = A(16x32 bf16) * B(32x16 bf16) + C(16x16 f32)
    return __builtin_amdgcn_wmma_f32_16x16x32_bf16(false, a, false, b, (short)0, c, false, false);
}

// A fragment: 16x32 bf16 tile from row-major source (lda elements).
// ISA: lane m=lane&15 holds row m; lanes 0-15 K in {0..7,16..23}, lanes 16-31 K in {8..15,24..31}.
static __device__ __forceinline__ v16bf load_a_frag(const bf16_t* A, int lda, int lane) {
    const int m  = lane & 15;
    const int kb = (lane >> 4) << 3;            // 0 or 8
    union { v16bf v; uint4 q[2]; } t;
    t.q[0] = *(const uint4*)(A + (size_t)m * lda + kb);
    t.q[1] = *(const uint4*)(A + (size_t)m * lda + 16 + kb);
    return t.v;
}

// B fragment where column n of B is contiguous in memory (B[k][n] = W[n][k], W row-major).
// ISA: lane n=lane&15 holds column n; lanes 0-15 K in {0..7,16..23}, lanes 16-31 K in {8..15,24..31}.
static __device__ __forceinline__ v16bf load_b_fragT(const bf16_t* W, int ldw, int lane) {
    const int n  = lane & 15;
    const int kb = (lane >> 4) << 3;            // 0 or 8
    const bf16_t* p = W + (size_t)n * ldw + kb;
    union { v16bf v; uint4 q[2]; } t;
    t.q[0] = *(const uint4*)p;
    t.q[1] = *(const uint4*)(p + 16);
    return t.v;
}

__global__ __launch_bounds__(256) void f32_to_bf16_kernel(const float* __restrict__ s,
                                                          bf16_t* __restrict__ d, int n) {
    const int i = blockIdx.x * 256 + threadIdx.x;
    if (i < n) d[i] = (bf16_t)s[i];
}

// ---------------------------------------------------------------------------
// C[M,N] = A[M,K] * W[N,K]^T (+bias). Register-blocked: 64x32 of C per wave
// (4 M-frags x 2 N-frags -> 8 WMMAs per 12 b128 loads), 8 waves -> 128x128/block.
// ---------------------------------------------------------------------------
__global__ __launch_bounds__(256) void gemm_bf16_wmma(
    const bf16_t* __restrict__ A, const bf16_t* __restrict__ W,
    float* __restrict__ Cf, bf16_t* __restrict__ Cb,
    const float* __restrict__ bias, int M, int N, int K)
{
    const int lane  = threadIdx.x & 31;
    const int wave  = threadIdx.x >> 5;
    const int nBlkN = N >> 7;
    const int bm = (blockIdx.x / nBlkN) << 7;
    const int bn = (blockIdx.x % nBlkN) << 7;
    const int tm = bm + ((wave >> 2) << 6);     // 64 rows per wave
    const int tn = bn + ((wave & 3) << 5);      // 32 cols per wave

    const v8f vzero = {0.f, 0.f, 0.f, 0.f, 0.f, 0.f, 0.f, 0.f};
    v8f acc[4][2];
    #pragma unroll
    for (int i = 0; i < 4; ++i)
        #pragma unroll
        for (int j = 0; j < 2; ++j) acc[i][j] = vzero;

    const bf16_t* pa = A + (size_t)tm * K;
    const bf16_t* pw = W + (size_t)tn * K;
    for (int kk = 0; kk < K; kk += 32) {
        __builtin_prefetch(pa + kk + 256, 0, 3);   // global_prefetch_b8
        __builtin_prefetch(pw + kk + 256, 0, 3);
        v16bf af[4], bf[2];
        #pragma unroll
        for (int i = 0; i < 4; ++i) af[i] = load_a_frag(pa + (size_t)(i * 16) * K + kk, K, lane);
        #pragma unroll
        for (int j = 0; j < 2; ++j) bf[j] = load_b_fragT(pw + (size_t)(j * 16) * K + kk, K, lane);
        #pragma unroll
        for (int i = 0; i < 4; ++i)
            #pragma unroll
            for (int j = 0; j < 2; ++j) acc[i][j] = wmma_bf16(af[i], bf[j], acc[i][j]);
    }

    #pragma unroll
    for (int j = 0; j < 2; ++j) {
        const int nn = tn + j * 16 + (lane & 15);
        const float bv = bias ? bias[nn] : 0.0f;
        #pragma unroll
        for (int i = 0; i < 4; ++i) {
            #pragma unroll
            for (int r = 0; r < 8; ++r) {
                const int mm = tm + i * 16 + ((lane >> 4) << 3) + r;  // M = r + 8*(lane>>4)
                const float v = acc[i][j][r] + bv;
                if (Cf) Cf[(size_t)mm * N + nn] = v;
                else    Cb[(size_t)mm * N + nn] = (bf16_t)v;
            }
        }
    }
}

// ---------------------------------------------------------------------------
// Tensor Data Mover: 2D bf16 tile global -> LDS, descriptor per ISA ch.8.
// ---------------------------------------------------------------------------
#if __has_builtin(__builtin_amdgcn_tensor_load_to_lds)
#define HAVE_TDM 1
typedef unsigned int u32x4 __attribute__((ext_vector_type(4)));
typedef int          i32x4 __attribute__((ext_vector_type(4)));
typedef int          i32x8 __attribute__((ext_vector_type(8)));
typedef __attribute__((address_space(3))) bf16_t lds_bf16;

// Build the invariant parts of the D# (group1 + lds_addr/count in group0).
static __device__ __forceinline__ void tdm_desc_2d_bf16(
    u32x4* g0, i32x8* g1, unsigned lds_off, int tile_d0, int tile_d1,
    int tensor_d0, int tensor_d1, long long stride0_elems)
{
    (*g0)[0] = 1u;                                             // count=1, user mode
    (*g0)[1] = lds_off;                                        // lds_addr (bytes)
    (*g0)[2] = 0u; (*g0)[3] = 0u;                              // patched per issue
    (*g1)[0] = (1 << 16);                                      // data_size=1 (2B), wg_mask=0
    (*g1)[1] = (int)(((unsigned)tensor_d0 & 0xFFFFu) << 16);   // tensor_dim0[15:0] @63:48
    (*g1)[2] = (int)(((unsigned)tensor_d0 >> 16) |
                     (((unsigned)tensor_d1 & 0xFFFFu) << 16)); // d0[31:16] | d1[15:0]
    (*g1)[3] = (int)(((unsigned)tensor_d1 >> 16) |
                     ((unsigned)tile_d0 << 16));               // d1[31:16] | tile_dim0
    (*g1)[4] = (int)((unsigned)tile_d1 & 0xFFFFu);             // tile_dim1, tile_dim2=0
    (*g1)[5] = (int)((unsigned long long)stride0_elems & 0xFFFFFFFFull);
    (*g1)[6] = (int)(((unsigned long long)stride0_elems >> 32) & 0xFFFFull);
    (*g1)[7] = 0;
}

static __device__ __forceinline__ void tdm_issue(u32x4 g0, i32x8 g1, unsigned long long ga)
{
    g0[2] = (unsigned)(ga & 0xFFFFFFFFu);                      // global_addr[31:0]
    g0[3] = (unsigned)((ga >> 32) & 0x01FFFFFFu) | (2u << 30); // addr[56:32] | type=2
    const i32x4 z4 = {0, 0, 0, 0};
#if __clang_major__ >= 23
    const i32x8 z8 = {0, 0, 0, 0, 0, 0, 0, 0};
    __builtin_amdgcn_tensor_load_to_lds(g0, g1, z4, z4, z8, 0);
#else
    __builtin_amdgcn_tensor_load_to_lds(g0, g1, z4, z4, 0);
#endif
}
#elif __has_builtin(__builtin_amdgcn_global_load_async_to_lds_b128)
#define HAVE_ASYNC 1
typedef __attribute__((address_space(1))) void gbl_void;
typedef __attribute__((address_space(3))) void lds_void;
#endif

// ---------------------------------------------------------------------------
// Flash attention: block = 8 waves / 128 queries; K staged via TDM (fallbacks:
// async-to-LDS, VGPR copies); V staged transposed (packed b32 stores).
// ---------------------------------------------------------------------------
__global__ __launch_bounds__(256) void flash_attn_bf16(
    const bf16_t* __restrict__ Qg, const bf16_t* __restrict__ Kg,
    const bf16_t* __restrict__ Vg, const float* __restrict__ head_w,
    bf16_t* __restrict__ Og)
{
    const int lane  = threadIdx.x & 31;
    const int wave  = threadIdx.x >> 5;                 // 0..7
    const int h     = blockIdx.y, b = blockIdx.z;
    const int qbase = (blockIdx.x << 7) + (wave << 4);  // 128 queries per block

    const size_t rowbase = (size_t)b * S_LEN;
    const bf16_t* Qp = Qg + (rowbase + qbase) * D_DIM + h * HD_DIM;
    const bf16_t* Kp = Kg + rowbase * D_DIM + h * HD_DIM;
    const bf16_t* Vp = Vg + rowbase * D_DIM + h * HD_DIM;

    __shared__ bf16_t ldsK[32 * HD_DIM];                // 8 KB, K tile row-major [kv][d]
    __shared__ bf16_t ldsVt[HD_DIM * LDV];              // 10 KB, V tile transposed [d][kv]
    __shared__ bf16_t ldsP[8][16 * 32];                 // 8 KB, per-wave P staging

    v16bf qf[4];                                        // Q 16x128 as 4 K-chunks
    #pragma unroll
    for (int c = 0; c < 4; ++c) qf[c] = load_a_frag(Qp + c * 32, D_DIM, lane);

    const v8f vzero = {0.f, 0.f, 0.f, 0.f, 0.f, 0.f, 0.f, 0.f};
    v8f o[8];
    #pragma unroll
    for (int t = 0; t < 8; ++t) o[t] = vzero;
    float rmax[8], rsum[8];
    #pragma unroll
    for (int r = 0; r < 8; ++r) { rmax[r] = -3.0e38f; rsum[r] = 0.f; }

    const float scale = 0.088388347648318447f;          // 1/sqrt(128)
    const int cn = lane & 15;

#if defined(HAVE_TDM)
    u32x4 kg0; i32x8 kg1;                               // descriptor built once
    tdm_desc_2d_bf16(&kg0, &kg1, (unsigned)(size_t)(lds_bf16*)&ldsK[0],
                     /*tile*/ HD_DIM, 32, /*tensor*/ D_DIM, S_LEN, D_DIM);
    const unsigned long long kaddr0 = (unsigned long long)(size_t)Kp;
#endif

    for (int kv = 0; kv < S_LEN; kv += 32) {
        __syncthreads();                                // previous tile fully consumed

        // ---- K tile: Tensor Data Mover (one issue per block; TDM ignores EXEC) ----
#if defined(HAVE_TDM)
        if (wave == 0)
            tdm_issue(kg0, kg1, kaddr0 + (unsigned long long)kv * (D_DIM * 2));
#elif defined(HAVE_ASYNC)
        for (int i = threadIdx.x; i < 512; i += 256) {
            const int row = i >> 4, cq = i & 15;
            __builtin_amdgcn_global_load_async_to_lds_b128(
                (gbl_void*)(Kp + (size_t)(kv + row) * D_DIM + cq * 8),
                (lds_void*)&ldsK[i * 8], 0, 0);
        }
#else
        for (int i = threadIdx.x; i < 512; i += 256) {
            const int row = i >> 4, cq = i & 15;
            ((uint4*)ldsK)[i] = *(const uint4*)(Kp + (size_t)(kv + row) * D_DIM + cq * 8);
        }
#endif
        // ---- V tile: row pairs -> transposed ldsVt[d][kv], packed b32 stores ----
        {
            const int i  = threadIdx.x;                 // 256 threads, one unit each
            const int rp = i >> 4;                      // row pair 0..15
            const int cq = i & 15;                      // d-chunk
            union { uint4 q; bf16_t e[8]; } t0, t1;
            t0.q = *(const uint4*)(Vp + (size_t)(kv + 2 * rp)     * D_DIM + cq * 8);
            t1.q = *(const uint4*)(Vp + (size_t)(kv + 2 * rp + 1) * D_DIM + cq * 8);
            unsigned* vt32 = (unsigned*)ldsVt;          // LDV/2 = 20 dwords per d-row
            #pragma unroll
            for (int u = 0; u < 8; ++u) {
                union { bf16_t e[2]; unsigned w; } pk;
                pk.e[0] = t0.e[u]; pk.e[1] = t1.e[u];
                vt32[(cq * 8 + u) * (LDV / 2) + rp] = pk.w;
            }
        }
#if defined(HAVE_TDM)
        if (wave == 0) __builtin_amdgcn_s_wait_tensorcnt(0);
#elif defined(HAVE_ASYNC)
        asm volatile("s_wait_asynccnt 0x0" ::: "memory");
#endif
        __syncthreads();

        // ---- Scores: S = Q * K^T, two 16-key tiles, 4 WMMAs each over Hd=128 ----
        v8f s0 = vzero, s1 = vzero;
        #pragma unroll
        for (int c = 0; c < 4; ++c) {
            s0 = wmma_bf16(qf[c], load_b_fragT(ldsK + c * 32, HD_DIM, lane), s0);
            s1 = wmma_bf16(qf[c], load_b_fragT(ldsK + 16 * HD_DIM + c * 32, HD_DIM, lane), s1);
        }

        // ---- Online softmax; lanes 0-15 hold rows r, lanes 16-31 rows 8+r ----
        bf16_t* pw = ldsP[wave];
        #pragma unroll
        for (int r = 0; r < 8; ++r) {
            const float v0 = s0[r] * scale, v1 = s1[r] * scale;
            float mx = fmaxf(v0, v1);
            #pragma unroll
            for (int off = 1; off < 16; off <<= 1) mx = fmaxf(mx, __shfl_xor(mx, off, 32));
            const float nm    = fmaxf(rmax[r], mx);
            const float alpha = __expf(rmax[r] - nm);
            const float p0    = __expf(v0 - nm);
            const float p1    = __expf(v1 - nm);
            const int   m     = ((lane >> 4) << 3) + r;
            pw[m * 32 + cn]      = (bf16_t)p0;          // P into A-fragment source layout
            pw[m * 32 + 16 + cn] = (bf16_t)p1;
            float rs = p0 + p1;
            #pragma unroll
            for (int off = 1; off < 16; off <<= 1) rs += __shfl_xor(rs, off, 32);
            rsum[r] = rsum[r] * alpha + rs;
            rmax[r] = nm;
            #pragma unroll
            for (int t = 0; t < 8; ++t) o[t][r] *= alpha;
        }

        // ---- O += P(16x32) * V(32x128): 8 WMMAs, B-frags are two ds b128 each ----
        const v16bf pf = load_a_frag(pw, 32, lane);
        #pragma unroll
        for (int t = 0; t < 8; ++t)
            o[t] = wmma_bf16(pf, load_b_fragT(ldsVt + (size_t)(t * 16) * LDV, LDV, lane), o[t]);
    }

    // ---- Normalize, apply per-head weight, store merged [B*S, D] bf16 ----
    const float hw = head_w[h];
    #pragma unroll
    for (int r = 0; r < 8; ++r) {
        const int m = ((lane >> 4) << 3) + r;
        const float inv = hw / rsum[r];
        #pragma unroll
        for (int t = 0; t < 8; ++t)
            Og[(rowbase + qbase + m) * D_DIM + h * HD_DIM + t * 16 + cn] = (bf16_t)(o[t][r] * inv);
    }
}

extern "C" void kernel_launch(void* const* d_in, const int* in_sizes, int n_in,
                              void* d_out, int out_size, void* d_ws, size_t ws_size,
                              hipStream_t stream) {
    (void)in_sizes; (void)n_in; (void)out_size; (void)ws_size;
    const float* x  = (const float*)d_in[0];
    const float* Wq = (const float*)d_in[1];
    const float* Wk = (const float*)d_in[2];
    const float* Wv = (const float*)d_in[3];
    const float* hw = (const float*)d_in[4];
    const float* Wu = (const float*)d_in[5];
    const float* bu = (const float*)d_in[6];
    float* out = (float*)d_out;

    const size_t M  = (size_t)B_SZ * S_LEN;        // 4096 rows
    const size_t XN = M * D_DIM;                   // 4,194,304 elems
    const size_t WN = (size_t)D_DIM * D_DIM;       // 1,048,576 elems

    char* ws = (char*)d_ws;
    bf16_t* xb  = (bf16_t*)ws; ws += XN * 2;
    bf16_t* wqb = (bf16_t*)ws; ws += WN * 2;
    bf16_t* wkb = (bf16_t*)ws; ws += WN * 2;
    bf16_t* wvb = (bf16_t*)ws; ws += WN * 2;
    bf16_t* wub = (bf16_t*)ws; ws += WN * 2;
    bf16_t* qb  = (bf16_t*)ws; ws += XN * 2;
    bf16_t* kb  = (bf16_t*)ws; ws += XN * 2;
    bf16_t* vb  = (bf16_t*)ws; ws += XN * 2;
    bf16_t* ab  = (bf16_t*)ws; ws += XN * 2;

    f32_to_bf16_kernel<<<(int)((XN + 255) / 256), 256, 0, stream>>>(x,  xb,  (int)XN);
    f32_to_bf16_kernel<<<(int)((WN + 255) / 256), 256, 0, stream>>>(Wq, wqb, (int)WN);
    f32_to_bf16_kernel<<<(int)((WN + 255) / 256), 256, 0, stream>>>(Wk, wkb, (int)WN);
    f32_to_bf16_kernel<<<(int)((WN + 255) / 256), 256, 0, stream>>>(Wv, wvb, (int)WN);
    f32_to_bf16_kernel<<<(int)((WN + 255) / 256), 256, 0, stream>>>(Wu, wub, (int)WN);

    const int gemmBlocks = (int)((M >> 7) * (D_DIM >> 7));   // 128x128 C per block
    gemm_bf16_wmma<<<gemmBlocks, 256, 0, stream>>>(xb, wqb, nullptr, qb, nullptr,
                                                   (int)M, D_DIM, D_DIM);
    gemm_bf16_wmma<<<gemmBlocks, 256, 0, stream>>>(xb, wkb, nullptr, kb, nullptr,
                                                   (int)M, D_DIM, D_DIM);
    gemm_bf16_wmma<<<gemmBlocks, 256, 0, stream>>>(xb, wvb, nullptr, vb, nullptr,
                                                   (int)M, D_DIM, D_DIM);

    dim3 ag(S_LEN / 128, H_NUM, B_SZ);
    flash_attn_bf16<<<ag, 256, 0, stream>>>(qb, kb, vb, hw, ab);

    gemm_bf16_wmma<<<gemmBlocks, 256, 0, stream>>>(ab, wub, out, nullptr, bu,
                                                   (int)M, D_DIM, D_DIM);
}
